// MultiheadAttention_42949673126
// MI455X (gfx1250) — matrix operand order repaired and checked by reading, not compiled, not run on hardware
//
#include <hip/hip_runtime.h>
#include <cstdint>

#define NN 50000
#define NE 800000
#define DIM 128
#define NH 8

typedef __attribute__((ext_vector_type(16))) __bf16 v16bf;
typedef __attribute__((ext_vector_type(8)))  float  v8f;

__device__ __forceinline__ float4 ldg4(const float* p) {
    return *reinterpret_cast<const float4*>(p);
}

__device__ __forceinline__ v16bf pack16(float4 x0, float4 x1, float4 x2, float4 x3) {
    v16bf r;
    r[0]=(__bf16)x0.x; r[1]=(__bf16)x0.y; r[2]=(__bf16)x0.z; r[3]=(__bf16)x0.w;
    r[4]=(__bf16)x1.x; r[5]=(__bf16)x1.y; r[6]=(__bf16)x1.z; r[7]=(__bf16)x1.w;
    r[8]=(__bf16)x2.x; r[9]=(__bf16)x2.y; r[10]=(__bf16)x2.z; r[11]=(__bf16)x2.w;
    r[12]=(__bf16)x3.x; r[13]=(__bf16)x3.y; r[14]=(__bf16)x3.z; r[15]=(__bf16)x3.w;
    return r;
}

// Hardware fp32 atomic add (no return -> STOREcnt). Guarantees
// GLOBAL_ATOMIC_ADD_F32 instead of a possible CAS loop.
__device__ __forceinline__ void atom_add(float* p, float v) {
    asm volatile("global_atomic_add_f32 %0, %1, off" :: "v"(p), "v"(v) : "memory");
}

// ---------------- zero-init d_out (agg accumulator) + w_sum ----------------
__global__ void zero_kernel(float* __restrict__ out, float* __restrict__ wsum) {
    int tid = blockIdx.x * blockDim.x + threadIdx.x;
    if (tid < NN * DIM) out[tid] = 0.0f;
    else if (tid < NN * DIM + NN * NH) wsum[tid - NN * DIM] = 0.0f;
}

// ---------------- fused QKV projection via bf16 WMMA ----------------
// One wave computes a 16-node x 64-output strip. 384 outputs = 6 strips.
// Strips 0-1 -> Q (scaled by 0.25), 2-3 -> K, 4-5 -> V.
__global__ void qkv_wmma_kernel(const float* __restrict__ emb,
                                const float* __restrict__ W,     // [384,128] row-major
                                const float* __restrict__ bias,  // [384]
                                float* __restrict__ Q,
                                float* __restrict__ K,
                                float* __restrict__ V) {
    const int lane  = threadIdx.x & 31;
    const int wid   = blockIdx.x * (blockDim.x >> 5) + (threadIdx.x >> 5);
    const int TOTAL = (NN / 16) * 6;
    if (wid >= TOTAL) return;

    const int rowTile = wid / 6;
    const int strip   = wid % 6;
    const int tileM   = rowTile * 16;
    const int tileN0  = strip * 64;
    const int m       = lane & 15;
    const int lhalf   = lane >> 4;

    v8f acc[4] = {v8f{}, v8f{}, v8f{}, v8f{}};

    for (int kb = 0; kb < DIM; kb += 32) {
        // --- A tile (16x32 bf16): lanes 0-15 hold K {0..7,16..23}, lanes 16-31 +8
        const float* ar = emb + (size_t)(tileM + m) * DIM + kb + lhalf * 8;
        v16bf a = pack16(ldg4(ar + 0), ldg4(ar + 4), ldg4(ar + 16), ldg4(ar + 20));

        // --- 4 B tiles (32x16 bf16 each): lanes 0-15 hold K {0..7,16..23}, lanes 16-31 +8
#pragma unroll
        for (int t = 0; t < 4; ++t) {
            const int n = tileN0 + t * 16 + m;           // output feature (row of W)
            const float* br = W + (size_t)n * DIM + kb + lhalf * 8;
            v16bf b = pack16(ldg4(br + 0), ldg4(br + 4), ldg4(br + 16), ldg4(br + 20));
            acc[t] = __builtin_amdgcn_wmma_f32_16x16x32_bf16(
                false, a, false, b, (short)0, acc[t], false, false);
        }
    }

    // Select destination matrix + scale (fold hd^-0.5 into Q)
    float scale;
    float* outbuf;
    int colOff;
    if (tileN0 < DIM)            { outbuf = Q; colOff = 0;       scale = 0.25f; }
    else if (tileN0 < 2 * DIM)   { outbuf = K; colOff = DIM;     scale = 1.0f; }
    else                         { outbuf = V; colOff = 2 * DIM; scale = 1.0f; }

    // D layout: VGPR r -> M = r + 8*lhalf, N = lane&15
#pragma unroll
    for (int t = 0; t < 4; ++t) {
        const int o   = tileN0 + t * 16 + m;
        const int col = o - colOff;
        const float bb = bias[o];
#pragma unroll
        for (int r = 0; r < 8; ++r) {
            const int node = tileM + r + lhalf * 8;
            outbuf[(size_t)node * DIM + col] = (acc[t][r] + bb) * scale;
        }
    }
}

// ---------------- edge phase: wave32 per edge, L2-resident gathers ----------------
__global__ void edge_kernel(const int* __restrict__ edges,
                            const float* __restrict__ Q,
                            const float* __restrict__ K,
                            const float* __restrict__ V,
                            float* __restrict__ agg,
                            float* __restrict__ wsum) {
    const int lane   = threadIdx.x & 31;
    const int wid    = blockIdx.x * (blockDim.x >> 5) + (threadIdx.x >> 5);
    const int nwaves = gridDim.x * (blockDim.x >> 5);
    const int h      = lane >> 2;  // 4 lanes x 4 floats = one 16-dim head

    for (int e = wid; e < NE; e += nwaves) {
        // Wave-uniform edge index: force scalar so the index loads and row
        // base addresses go through the scalar path.
        const int eu  = __builtin_amdgcn_readfirstlane(e);
        const int src = __builtin_amdgcn_readfirstlane(edges[eu]);
        const int dst = __builtin_amdgcn_readfirstlane(edges[NE + eu]);

        const float4 q = ldg4(Q + (size_t)src * DIM + lane * 4);  // already scaled
        const float4 k = ldg4(K + (size_t)dst * DIM + lane * 4);
        const float4 v = ldg4(V + (size_t)dst * DIM + lane * 4);

        float p = q.x * k.x + q.y * k.y + q.z * k.z + q.w * k.w;
        p += __shfl_xor(p, 1, 32);
        p += __shfl_xor(p, 2, 32);       // sum over the 4-lane head group
        const float w = __expf(p);

        if ((lane & 3) == 0)
            atom_add(&wsum[(size_t)src * NH + h], w);

        float* ab = agg + (size_t)src * DIM + lane * 4;
        atom_add(ab + 0, w * v.x);
        atom_add(ab + 1, w * v.y);
        atom_add(ab + 2, w * v.z);
        atom_add(ab + 3, w * v.w);
    }
}

// ---------------- normalize in place ----------------
__global__ void norm_kernel(float* __restrict__ out, const float* __restrict__ wsum) {
    int tid = blockIdx.x * blockDim.x + threadIdx.x;
    if (tid >= NN * DIM) return;
    const int node = tid >> 7;
    const int h    = (tid >> 4) & 7;
    out[tid] = out[tid] / (wsum[(size_t)node * NH + h] + 1e-20f);
}

extern "C" void kernel_launch(void* const* d_in, const int* in_sizes, int n_in,
                              void* d_out, int out_size, void* d_ws, size_t ws_size,
                              hipStream_t stream) {
    const float* emb  = (const float*)d_in[0];
    const float* W    = (const float*)d_in[1];
    const float* bias = (const float*)d_in[2];
    const int*   edges = (const int*)d_in[3];
    float* out = (float*)d_out;

    float* Q    = (float*)d_ws;
    float* K    = Q + (size_t)NN * DIM;
    float* V    = K + (size_t)NN * DIM;
    float* wsum = V + (size_t)NN * DIM;   // NN*NH floats; total ws = 78.4 MB

    {
        int total = NN * DIM + NN * NH;
        zero_kernel<<<(total + 255) / 256, 256, 0, stream>>>(out, wsum);
    }
    {
        int waves  = (NN / 16) * 6;             // 18750 waves
        int blocks = (waves + 3) / 4;           // 4 waves / block
        qkv_wmma_kernel<<<blocks, 128, 0, stream>>>(emb, W, bias, Q, K, V);
    }
    edge_kernel<<<2048, 256, 0, stream>>>(edges, Q, K, V, out, wsum);
    {
        int total = NN * DIM;
        norm_kernel<<<(total + 255) / 256, 256, 0, stream>>>(out, wsum);
    }
}
